// NonlinearMixtureMobile_35381940584884
// MI455X (gfx1250) — matrix operand order repaired and checked by reading, not compiled, not run on hardware
//
#include <hip/hip_runtime.h>
#include <hip/hip_fp16.h>

// ---- types ----
typedef _Float16 h8  __attribute__((ext_vector_type(8)));
typedef _Float16 h16 __attribute__((ext_vector_type(16)));
typedef float    f8  __attribute__((ext_vector_type(8)));
typedef float    f4  __attribute__((ext_vector_type(4)));

#define B_N 1024
#define E_N 8
#define D_N 3072   // 3*32*32

// workspace layout (float offsets). Total ~10.1 MB.
#define WS_SS    0                       // [1024*8] select_softmax
#define WS_NORM  8192                    // [1024] squared norms
#define WS_GATE  9216                    // [1024] gate values
#define WS_IDX   10240                   // [1024] int expert index
#define WS_GMAX  11264                   // [1] max softmax value (float bits via atomicMax int)
#define WS_XH    12288                   // [1024*3072] _Float16 (occupies 1572864 float slots)
#define WS_G     (12288 + 1572864)       // [1024*1024] float Gram matrix (symmetric)

// ---------------------------------------------------------------------------
__global__ void init_kernel(float* ws) {
  if (threadIdx.x == 0) ((int*)(ws + WS_GMAX))[0] = 0;   // 0.0f bits
}

// ---------------------------------------------------------------------------
__global__ __launch_bounds__(256) void cvt_kernel(const float* __restrict__ x,
                                                  _Float16* __restrict__ xh) {
  int i = blockIdx.x * 256 + threadIdx.x;
  xh[i] = (_Float16)x[i];
}

// ---------------------------------------------------------------------------
// Router: select[b,e] = sum_{c,h,w} x * rw[e,c,h%4,w%4] + rb[e]; softmax; also ||x_b||^2.
__global__ __launch_bounds__(256) void router_kernel(const float* __restrict__ x,
                                                     const float* __restrict__ rw,
                                                     const float* __restrict__ rb,
                                                     float* __restrict__ ws) {
  __shared__ float rwS[384];       // 8*3*4*4
  __shared__ float part[9][8];
  const int b = blockIdx.x, t = threadIdx.x;
  if (t < 384) rwS[t] = rw[t];
  __syncthreads();

  float acc[E_N] = {0,0,0,0,0,0,0,0};
  float nrm = 0.0f;
  const float* xb = x + (size_t)b * D_N;
  for (int i = t; i < D_N; i += 256) {
    float xv = xb[i];
    nrm += xv * xv;
    int c = i >> 10, rem = i & 1023, h = rem >> 5, w = rem & 31;
    int base = c * 16 + (h & 3) * 4 + (w & 3);
#pragma unroll
    for (int e = 0; e < E_N; e++) acc[e] += xv * rwS[e * 48 + base];
  }
  const int lane = t & 31, wid = t >> 5;
#pragma unroll
  for (int q = 0; q < E_N; q++) {
    float p = acc[q];
    for (int off = 16; off; off >>= 1) p += __shfl_down(p, off, 32);
    if (lane == 0) part[q][wid] = p;
  }
  {
    float p = nrm;
    for (int off = 16; off; off >>= 1) p += __shfl_down(p, off, 32);
    if (lane == 0) part[8][wid] = p;
  }
  __syncthreads();
  if (t == 0) {
    float sel[E_N];
    float mx = -1e30f;
#pragma unroll
    for (int e = 0; e < E_N; e++) {
      float s = rb[e];
      for (int w = 0; w < 8; w++) s += part[e][w];
      sel[e] = s;
      mx = fmaxf(mx, s);
    }
    float se[E_N], ssum = 0.0f;
#pragma unroll
    for (int e = 0; e < E_N; e++) { se[e] = expf(sel[e] - mx); ssum += se[e]; }
    float smax = 0.0f;
#pragma unroll
    for (int e = 0; e < E_N; e++) {
      float v = se[e] / ssum;
      ws[WS_SS + b * E_N + e] = v;
      smax = fmaxf(smax, v);
    }
    float n2 = 0.0f;
    for (int w = 0; w < 8; w++) n2 += part[8][w];
    ws[WS_NORM + b] = n2;
    atomicMax((int*)(ws + WS_GMAX), __float_as_int(smax));  // all values > 0
  }
}

// ---------------------------------------------------------------------------
// Gram matrix via WMMA: S = Xh * Xh^T (f16 in, f32 acc), then G = exp(-max(d,0)).
// Each wave computes a 16x64 strip (4 N-tiles reusing one A operand per k-step).
// 64 x 16 strip-tasks = 1024 wave-tasks; 8 waves/block -> 128 blocks.
__global__ __launch_bounds__(256) void gram_kernel(const _Float16* __restrict__ Xh,
                                                   const float* __restrict__ norms,
                                                   float* __restrict__ G) {
  const int wid  = threadIdx.x >> 5;
  const int lane = threadIdx.x & 31;
  const int task = blockIdx.x * 8 + wid;
  const int ti = task >> 4;                 // 0..63  (M block of 16)
  const int tg = task & 15;                 // 0..15  (N strip of 64)
  const int i0 = ti * 16, j0 = tg * 64;
  const int half = lane >> 4, r = lane & 15;

  const _Float16* arow  = Xh + (size_t)(i0 + r) * D_N;
  const _Float16* brow0 = Xh + (size_t)(j0 + r) * D_N + 16 * half;

  f8 acc0 = {0.f,0.f,0.f,0.f,0.f,0.f,0.f,0.f};
  f8 acc1 = acc0, acc2 = acc0, acc3 = acc0;

#pragma unroll 2
  for (int k = 0; k < D_N; k += 32) {
    // A (16x32 f16): lane holds K = 16*half + 0..15
    h16 A = *(const h16*)(arow + k + 16 * half);
    // B (32x16 f16): lane n=r holds K = 16*half + 0..15; 4 N-tiles
    h16 B0 = *(const h16*)(brow0 + k);
    h16 B1 = *(const h16*)(brow0 + (size_t)16 * D_N + k);
    h16 B2 = *(const h16*)(brow0 + (size_t)32 * D_N + k);
    h16 B3 = *(const h16*)(brow0 + (size_t)48 * D_N + k);
    acc0 = __builtin_amdgcn_wmma_f32_16x16x32_f16(false, A, false, B0, (short)0, acc0, false, false);
    acc1 = __builtin_amdgcn_wmma_f32_16x16x32_f16(false, A, false, B1, (short)0, acc1, false, false);
    acc2 = __builtin_amdgcn_wmma_f32_16x16x32_f16(false, A, false, B2, (short)0, acc2, false, false);
    acc3 = __builtin_amdgcn_wmma_f32_16x16x32_f16(false, A, false, B3, (short)0, acc3, false, false);
  }

  // C/D layout: vgpr q holds (M = q + 8*half, N = r)
  float ni[8];
#pragma unroll
  for (int q = 0; q < 8; q++) ni[q] = norms[i0 + q + 8 * half];
#pragma unroll
  for (int n = 0; n < 4; n++) {
    const f8& a = (n == 0) ? acc0 : (n == 1) ? acc1 : (n == 2) ? acc2 : acc3;
    int gj = j0 + n * 16 + r;
    float nj = norms[gj];
#pragma unroll
    for (int q = 0; q < 8; q++) {
      int gi = i0 + q + 8 * half;
      float d = ni[q] + nj - 2.0f * a[q];
      d = fmaxf(d, 0.0f);
      if (gi == gj) d = 0.0f;              // exact diagonal (matches f32 reference)
      G[(size_t)gi * B_N + gj] = expf(-d); // KHP = 1
    }
  }
}

// ---------------------------------------------------------------------------
// GW-entropic OT solver: single block, 1024 threads (thread j owns column j).
// pi stored i-major (pis[i][e]) so the T = pi@G loop needs only 2 ds_load_b128
// broadcasts per i; G is symmetric so thread j streams row j with b128 loads.
__global__ __launch_bounds__(1024) void solver_kernel(const float* __restrict__ ws,
                                                      const float* __restrict__ G,
                                                      float* __restrict__ gate,
                                                      int* __restrict__ idxO,
                                                      float* __restrict__ outSel,
                                                      float* __restrict__ outLoss,
                                                      float* __restrict__ outCnt) {
  __shared__ float pis[B_N][E_N];   // 32 KB, i-major
  __shared__ float part[E_N][32];
  __shared__ float u8[E_N];
  __shared__ float mxS;
  __shared__ int   cnt[E_N];
  __shared__ float proxS[E_N];

  const int j = threadIdx.x;
  const int lane = j & 31, wid = j >> 5;
  const float* ss = ws + WS_SS;

  const float invm = 1.0f / (ws[WS_GMAX] + 1e-9f);
  float Cc[E_N], ssj[E_N];
#pragma unroll
  for (int e = 0; e < E_N; e++) {
    ssj[e] = ss[j * E_N + e];
    Cc[e] = -ssj[e] * invm;           // C = -ss.T / (max|ss| + 1e-9)
  }
#pragma unroll
  for (int e = 0; e < E_N; e++) pis[j][e] = 0.125f;   // outer(mu,nu)/sum(nu)
  if (j < E_N) cnt[j] = 0;
  __syncthreads();

  const f4* __restrict__ Grow = (const f4*)(G + (size_t)j * B_N);  // row j == col j

  for (int outer = 0; outer < 25; outer++) {
    // T = pi @ G   (G_n = I)
    float t[E_N] = {0,0,0,0,0,0,0,0};
    for (int i4 = 0; i4 < B_N / 4; i4++) {
      f4 g = Grow[i4];
      const f4* pp = (const f4*)(&pis[i4 * 4][0]);
#pragma unroll
      for (int q = 0; q < 4; q++) {
        f4 p0 = pp[q * 2 + 0];
        f4 p1 = pp[q * 2 + 1];
        float gg = g[q];
        t[0] += p0[0] * gg; t[1] += p0[1] * gg; t[2] += p0[2] * gg; t[3] += p0[3] * gg;
        t[4] += p1[0] * gg; t[5] += p1[1] * gg; t[6] += p1[2] * gg; t[7] += p1[3] * gg;
      }
    }
    float ce[E_N];
    float mx = 0.0f;
#pragma unroll
    for (int e = 0; e < E_N; e++) {
      ce[e] = Cc[e] - t[e];           // LDA3 = 1
      mx = fmaxf(mx, fabsf(ce[e]));
    }
    for (int off = 16; off; off >>= 1) mx = fmaxf(mx, __shfl_down(mx, off, 32));
    if (lane == 0) part[0][wid] = mx;
    __syncthreads();
    if (j == 0) {
      float m2 = 0.0f;
      for (int w = 0; w < 32; w++) m2 = fmaxf(m2, part[0][w]);
      mxS = m2;
    }
    __syncthreads();
    const float sc = 1.0f / ((mxS + 1e-9f) * 0.1f);   // /(max+eps) then /LDA
    float Ke[E_N];
#pragma unroll
    for (int e = 0; e < E_N; e++) Ke[e] = expf(-ce[e] * sc);
    __syncthreads();

    // Sinkhorn: 5x (u,v) + final u
    float vj = 1.0f;
    for (int it = 0; it < 6; it++) {
#pragma unroll
      for (int e = 0; e < E_N; e++) {
        float p = Ke[e] * vj;
        for (int off = 16; off; off >>= 1) p += __shfl_down(p, off, 32);
        if (lane == 0) part[e][wid] = p;
      }
      __syncthreads();
      if (j < E_N) {
        float s = 0.0f;
        for (int w = 0; w < 32; w++) s += part[j][w];
        u8[j] = 128.0f / (s + 1e-9f);     // mu = m/n = 1024/8
      }
      __syncthreads();
      if (it < 5) {
        float s = 0.0f;
#pragma unroll
        for (int e = 0; e < E_N; e++) s += Ke[e] * u8[e];
        vj = 1.0f / (s + 1e-9f);          // nu = 1
      }
    }
#pragma unroll
    for (int e = 0; e < E_N; e++) pis[j][e] = u8[e] * Ke[e] * vj;
    __syncthreads();
  }

  // argmax over experts (first max, like jnp.argmax), gate, one-hot, counts
  int best = 0;
  float bv = pis[j][0];
  float g = ssj[0];
#pragma unroll
  for (int e = 1; e < E_N; e++) {
    float p = pis[j][e];
    if (p > bv) { bv = p; best = e; g = ssj[e]; }
  }
  gate[j] = g;
  idxO[j] = best;
#pragma unroll
  for (int e = 0; e < E_N; e++) outSel[j * E_N + e] = (e == best) ? 1.0f : 0.0f;
  atomicAdd(&cnt[best], 1);

  // density_proxy[e] = mean_j ss[j,e]
#pragma unroll
  for (int e = 0; e < E_N; e++) {
    float p = ssj[e];
    for (int off = 16; off; off >>= 1) p += __shfl_down(p, off, 32);
    if (lane == 0) part[e][wid] = p;
  }
  __syncthreads();
  if (j < E_N) {
    float s = 0.0f;
    for (int w = 0; w < 32; w++) s += part[j][w];
    proxS[j] = s * (1.0f / 1024.0f);
    outCnt[j] = (float)cnt[j];
  }
  __syncthreads();
  if (j == 0) {
    float L = 0.0f;
    for (int e = 0; e < E_N; e++) L += proxS[e] * ((float)cnt[e] * (1.0f / 1024.0f));
    *outLoss = L * (64.0f / 8.0f);        // mean over E, * E^2
  }
}

// ---------------------------------------------------------------------------
// Per-sample forward through its selected expert, scaled by gate.
__global__ __launch_bounds__(256) void expert_kernel(const float* __restrict__ x,
    const float* __restrict__ w1, const float* __restrict__ b1,
    const float* __restrict__ wd, const float* __restrict__ bd,
    const float* __restrict__ wp, const float* __restrict__ bp,
    const float* __restrict__ wf, const float* __restrict__ bf,
    const float* __restrict__ gate, const int* __restrict__ idx,
    float* __restrict__ out) {
  __shared__ float xp[3][34][34];     // zero-padded input
  __shared__ float w1s[432], b1s[16], wds[144], bds[16];
  __shared__ float wps[512], bps[32], wfs[320], bfs[16];
  __shared__ float h1b[4][33 * 33];   // 4-channel chunk, +1 high-side pad
  __shared__ float h2b[16][256];      // after depthwise stride-2
  __shared__ float partm[32][8];
  __shared__ float meanS[32];

  const int b = blockIdx.x, t = threadIdx.x;
  const int e = idx[b];
  const float gv = gate[b];

  for (int i = t; i < 432; i += 256) w1s[i] = w1[e * 432 + i];
  if (t < 16) b1s[t] = b1[e * 16 + t];
  if (t < 144) wds[t] = wd[e * 144 + t];
  if (t < 16) bds[t] = bd[e * 16 + t];
  for (int i = t; i < 512; i += 256) wps[i] = wp[e * 512 + i];
  if (t < 32) bps[t] = bp[e * 32 + t];
  for (int i = t; i < 320; i += 256) wfs[i] = wf[e * 320 + i];
  if (t < 10) bfs[t] = bf[e * 10 + t];
  for (int i = t; i < 3 * 34 * 34; i += 256) ((float*)xp)[i] = 0.0f;
  __syncthreads();
  const float* xb = x + (size_t)b * D_N;
  for (int i = t; i < D_N; i += 256) {
    int c = i >> 10, rem = i & 1023, h = rem >> 5, w = rem & 31;
    xp[c][h + 1][w + 1] = xb[i];
  }
  __syncthreads();

  for (int cc = 0; cc < 4; cc++) {
    for (int i = t; i < 4 * 1089; i += 256) ((float*)h1b)[i] = 0.0f;
    __syncthreads();
    // conv1 3->16, 3x3 SAME + ReLU6, for channels cc*4 .. cc*4+3
    for (int i = t; i < 4096; i += 256) {
      int ch = i >> 10, rem = i & 1023, oh = rem >> 5, ow = rem & 31;
      int oc = cc * 4 + ch;
      float acc = b1s[oc];
#pragma unroll
      for (int ic = 0; ic < 3; ic++)
#pragma unroll
        for (int kh = 0; kh < 3; kh++)
#pragma unroll
          for (int kw = 0; kw < 3; kw++)
            acc += xp[ic][oh + kh][ow + kw] * w1s[oc * 27 + ic * 9 + kh * 3 + kw];
      acc = fminf(fmaxf(acc, 0.0f), 6.0f);
      h1b[ch][oh * 33 + ow] = acc;
    }
    __syncthreads();
    // depthwise 3x3 stride 2 SAME (pad_lo=0, pad_hi=1) + ReLU6
    for (int i = t; i < 1024; i += 256) {
      int ch = i >> 8, rem = i & 255, oh = rem >> 4, ow = rem & 15;
      int c = cc * 4 + ch;
      float acc = bds[c];
#pragma unroll
      for (int kh = 0; kh < 3; kh++)
#pragma unroll
        for (int kw = 0; kw < 3; kw++)
          acc += h1b[ch][(2 * oh + kh) * 33 + (2 * ow + kw)] * wds[c * 9 + kh * 3 + kw];
      acc = fminf(fmaxf(acc, 0.0f), 6.0f);
      h2b[c][oh * 16 + ow] = acc;
    }
    __syncthreads();
  }

  // pointwise 16->32 + ReLU6, then spatial mean
  const int lane = t & 31, wid = t >> 5;
  float po[32];
#pragma unroll
  for (int o = 0; o < 32; o++) {
    float acc = bps[o];
#pragma unroll
    for (int c = 0; c < 16; c++) acc += h2b[c][t] * wps[o * 16 + c];
    po[o] = fminf(fmaxf(acc, 0.0f), 6.0f);
  }
#pragma unroll
  for (int o = 0; o < 32; o++) {
    float s = po[o];
    for (int off = 16; off; off >>= 1) s += __shfl_down(s, off, 32);
    if (lane == 0) partm[o][wid] = s;
  }
  __syncthreads();
  if (t < 32) {
    float s = 0.0f;
#pragma unroll
    for (int w = 0; w < 8; w++) s += partm[t][w];
    meanS[t] = s * (1.0f / 256.0f);
  }
  __syncthreads();
  if (t < 10) {
    float r = bfs[t];
#pragma unroll
    for (int o = 0; o < 32; o++) r += meanS[o] * wfs[t * 32 + o];
    out[(size_t)b * 10 + t] = gv * r;
  }
}

// ---------------------------------------------------------------------------
extern "C" void kernel_launch(void* const* d_in, const int* in_sizes, int n_in,
                              void* d_out, int out_size, void* d_ws, size_t ws_size,
                              hipStream_t stream) {
  (void)in_sizes; (void)n_in; (void)out_size; (void)ws_size;
  const float* x  = (const float*)d_in[0];
  const float* rw = (const float*)d_in[1];
  const float* rb = (const float*)d_in[2];
  const float* w1 = (const float*)d_in[3];
  const float* b1 = (const float*)d_in[4];
  const float* wd = (const float*)d_in[5];
  const float* bd = (const float*)d_in[6];
  const float* wp = (const float*)d_in[7];
  const float* bp = (const float*)d_in[8];
  const float* wf = (const float*)d_in[9];
  const float* bf = (const float*)d_in[10];
  float* out = (float*)d_out;
  float* ws  = (float*)d_ws;
  _Float16* Xh = (_Float16*)(ws + WS_XH);
  float* G = ws + WS_G;

  init_kernel<<<1, 1, 0, stream>>>(ws);
  cvt_kernel<<<(B_N * D_N) / 256, 256, 0, stream>>>(x, Xh);
  router_kernel<<<B_N, 256, 0, stream>>>(x, rw, rb, ws);
  gram_kernel<<<128, 256, 0, stream>>>(Xh, ws + WS_NORM, G);
  solver_kernel<<<1, 1024, 0, stream>>>(ws, G, ws + WS_GATE, (int*)(ws + WS_IDX),
                                        out + 10240, out + 18432, out + 18433);
  expert_kernel<<<B_N, 256, 0, stream>>>(x, w1, b1, wd, bd, wp, bp, wf, bf,
                                         ws + WS_GATE, (const int*)(ws + WS_IDX), out);
}
